// ImportanceRenderer_77670188581447
// MI455X (gfx1250) — matrix operand order repaired and checked by reading, not compiled, not run on hardware
//
#include <hip/hip_runtime.h>
#include <math.h>

typedef __attribute__((ext_vector_type(16))) _Float16     v16h;
typedef __attribute__((ext_vector_type(8)))  float        v8f;
typedef __attribute__((ext_vector_type(4)))  unsigned int v4u;
typedef __attribute__((ext_vector_type(8)))  int          v8i;
typedef __attribute__((ext_vector_type(4)))  int          v4i;

#define RAY_START 0.05f
#define RAY_END   0.95f

#if __has_builtin(__builtin_amdgcn_tensor_load_to_lds) && __has_builtin(__builtin_amdgcn_s_wait_tensorcnt)
#define HAVE_TDM 1
#else
#define HAVE_TDM 0
#endif

__device__ __forceinline__ float softplusf(float x) {
    return (x > 20.f) ? x : log1pf(expf(x));
}
__device__ __forceinline__ float sigmoidf_(float x) {
    return 1.f / (1.f + expf(-x));
}

#if HAVE_TDM
// 1-D TDM copy: nelem f32 from global -> LDS byte offset. Descriptor per
// CDNA5 ISA: group0 = {flags, lds_addr, gaddr_lo,
// gaddr_hi|type=2}; group1 packs data_size=4B, tensor_dim0=tile_dim0=nelem,
// tensor_dim1=tile_dim1=1, stride0=nelem. Groups 2/3/4 unused (zeros).
// This toolchain exposes the 6-arg builtin form:
//   (uint32x4 g0, int32x8 g1, int32x4, int32x4, int32x8, i32 cpol)
__device__ __forceinline__ void tdm_copy_1d_f32(const float* gsrc,
                                                unsigned lds_byte_off, int nelem)
{
    unsigned long long ga = (unsigned long long)(uintptr_t)gsrc;
    v4u g0;
    g0[0] = 1u;                                          // count=1 (valid user D#)
    g0[1] = lds_byte_off;                                // lds_addr
    g0[2] = (unsigned)(ga & 0xFFFFFFFFu);                // global_addr[31:0]
    g0[3] = (unsigned)((ga >> 32) & 0x1FFFFFFu) | (2u << 30); // addr[56:32] | type=2
    v8i g1;
    g1[0] = (int)(2u << 16);                             // data_size=2 (4 bytes)
    g1[1] = (int)(((unsigned)nelem & 0xFFFFu) << 16);    // tensor_dim0 lo16
    g1[2] = (int)((((unsigned)nelem >> 16) & 0xFFFFu) | (1u << 16)); // td0 hi | td1=1
    g1[3] = (int)(((unsigned)nelem & 0xFFFFu) << 16);    // td1 hi=0 | tile_dim0
    g1[4] = 1;                                           // tile_dim1=1, tile_dim2=0
    g1[5] = nelem;                                       // tensor_dim0_stride lo32
    g1[6] = 0;
    g1[7] = 0;
    v4i z4 = {0, 0, 0, 0};
    v8i z8 = {0, 0, 0, 0, 0, 0, 0, 0};
    __builtin_amdgcn_tensor_load_to_lds(g0, g1, z4, z4, z8, 0);
}
#endif

// ---------------------------------------------------------------------------
// Kernel 1/4: shade points. Tri-plane trilinear gather -> WMMA MLP decoder.
// One wave = 32 points = two 16-row WMMA tiles. Weights staged block-wide in
// LDS via TDM (overlapped with the gather phase). mode==0: coarse (depth from
// noise, stored); mode==1: fine (depth read from depths[]).
// ---------------------------------------------------------------------------
__global__ __launch_bounds__(256)
void shade_points(const float* __restrict__ planes,
                  const float* __restrict__ w1, const float* __restrict__ b1,
                  const float* __restrict__ w2, const float* __restrict__ b2,
                  const float* __restrict__ ro, const float* __restrict__ rd,
                  const float* __restrict__ noise,
                  float* __restrict__ depths,   // P (out if mode==0, in if mode==1)
                  float* __restrict__ rgba,     // P*4 out: [sigma_raw, r, g, b]
                  int mode, int P, int R, int S)
{
    // [0..511]=w1(8x64) [512..575]=b1 [576..831]=w2(64x4) [832..835]=b2
    __shared__ float    lds_wts[896];
    __shared__ _Float16 lds_h[8][32 * 64];   // per-wave staging (feats, then h)

    const int tid  = threadIdx.x;
    const int lane = tid & 31;
    const int wave = tid >> 5;
    const int tile = blockIdx.x * 8 + wave;
    const int p    = tile * 32 + lane;
    const bool valid = (p < P);
    const int pc = valid ? p : (P - 1);

    // --- kick off block-wide weight staging (DMA overlaps the gather) -------
#if HAVE_TDM
    if (tid == 0) {
        const unsigned base = (unsigned)(uintptr_t)(&lds_wts[0]);
        tdm_copy_1d_f32(w1, base,        512);
        tdm_copy_1d_f32(b1, base + 2048,  64);
        tdm_copy_1d_f32(w2, base + 2304, 256);
        tdm_copy_1d_f32(b2, base + 3328,   4);
    }
#else
    {
        for (int i = tid; i < 512; i += 256) lds_wts[i]       = w1[i];
        if (tid < 64)                        lds_wts[512+tid] = b1[tid];
        for (int i = tid; i < 256; i += 256) lds_wts[576+i]   = w2[i];
        if (tid < 4)                         lds_wts[832+tid] = b2[tid];
    }
#endif

    const int s  = pc % S;
    const int br = pc / S;        // b*R + ray
    const int b  = br / R;

    const float delta = (RAY_END - RAY_START) / (float)(S - 1);
    float depth;
    if (mode == 0) depth = RAY_START + (float)s * delta + noise[pc] * delta;
    else           depth = depths[pc];

    const float ox = ro[br * 3 + 0], oy = ro[br * 3 + 1], oz = ro[br * 3 + 2];
    const float ddx = rd[br * 3 + 0], ddy = rd[br * 3 + 1], ddz = rd[br * 3 + 2];
    // 2/BOX_WARP == 1.0
    const float cx = ox + depth * ddx;
    const float cy = oy + depth * ddy;
    const float cz = oz + depth * ddz;

    // --- tri-plane trilinear sampling, mean over 3 planes -------------------
    float feats[8];
#pragma unroll
    for (int k = 0; k < 8; ++k) feats[k] = 0.f;

#pragma unroll
    for (int pl = 0; pl < 3; ++pl) {
        float gx, gy, gz;
        if (pl == 0)      { gx = cx; gy = cy; gz = cz; }   // inv(identity)
        else if (pl == 1) { gx = cx; gy = cz; gz = cy; }   // swap y/z
        else              { gx = cy; gy = cz; gz = cx; }   // cyclic
        const float x = (gx + 1.f) * 128.f - 0.5f;   // W=256
        const float y = (gy + 1.f) * 128.f - 0.5f;   // H=256
        const float z = (gz + 1.f) * 2.f   - 0.5f;   // D=4
        const float x0f = floorf(x), y0f = floorf(y), z0f = floorf(z);
        const float fx = x - x0f, fy = y - y0f, fz = z - z0f;
        const int x0 = (int)x0f, y0 = (int)y0f, z0 = (int)z0f;
        const float* vbase = planes + (size_t)(b * 3 + pl) * 32 * 65536;
#pragma unroll
        for (int dzi = 0; dzi < 2; ++dzi) {
            const int zi = z0 + dzi;
            if (zi < 0 || zi >= 4) continue;
            const float wz = dzi ? fz : 1.f - fz;
#pragma unroll
            for (int dyi = 0; dyi < 2; ++dyi) {
                const int yi = y0 + dyi;
                if (yi < 0 || yi >= 256) continue;
                const float wy = dyi ? fy : 1.f - fy;
#pragma unroll
                for (int dxi = 0; dxi < 2; ++dxi) {
                    const int xi = x0 + dxi;
                    if (xi < 0 || xi >= 256) continue;
                    const float w = wz * wy * (dxi ? fx : 1.f - fx);
                    const float* cp = vbase + (size_t)zi * 65536 + yi * 256 + xi;
#pragma unroll
                    for (int c = 0; c < 8; ++c)
                        feats[c] += w * cp[(size_t)c * 262144];  // C0 stride 4*65536
                }
            }
        }
    }
#pragma unroll
    for (int k = 0; k < 8; ++k) feats[k] *= (1.f / 3.f);

    // stage feats in LDS (first 32*8 halfs of this wave's buffer)
    _Float16* ldsF = &lds_h[wave][0];
#pragma unroll
    for (int k = 0; k < 8; ++k) ldsF[lane * 8 + k] = (_Float16)feats[k];

#if HAVE_TDM
    if (tid < 32) __builtin_amdgcn_s_wait_tensorcnt(0);  // wave 0 drains its DMA
#endif
    __syncthreads();   // feats + staged weights visible block-wide

    // --- layer 1: A = feats (2 M-tiles x K=32 padded), B = w1 (4 N-tiles) ---
    const int n15 = lane & 15;
    const int hi  = lane >> 4;
    v16h a0 = {}, a1 = {};
    if (lane < 16) {
#pragma unroll
        for (int k = 0; k < 8; ++k) {
            a0[k] = ldsF[n15 * 8 + k];          // rows 0..15, K=0..7
            a1[k] = ldsF[(16 + n15) * 8 + k];   // rows 16..31
        }
    } // lanes>=16 carry K=8..15 / 16..31 -> all zero (K is only 8 wide)

    v16h bw[4];
#pragma unroll
    for (int t = 0; t < 4; ++t) {
        v16h bb = {};
        if (lane < 16) {                         // lane<16: col n, K=0..15
#pragma unroll
            for (int k = 0; k < 8; ++k)
                bb[k] = (_Float16)lds_wts[k * 64 + t * 16 + n15];
        }                                        // lane>=16: K=16..31 -> zero
        bw[t] = bb;
    }

    v8f h0[4], h1[4];
#pragma unroll
    for (int t = 0; t < 4; ++t) {
        v8f c0 = {}, c1 = {};
        h0[t] = __builtin_amdgcn_wmma_f32_16x16x32_f16(false, a0, false, bw[t],
                                                       (short)0, c0, false, false);
        h1[t] = __builtin_amdgcn_wmma_f32_16x16x32_f16(false, a1, false, bw[t],
                                                       (short)0, c1, false, false);
    }

    // bias + softplus, scatter h (32x64) to LDS in row-major halfs
#pragma unroll
    for (int t = 0; t < 4; ++t) {
        const float bias = lds_wts[512 + t * 16 + n15];
#pragma unroll
        for (int r = 0; r < 8; ++r) {
            const int m = r + 8 * hi;
            lds_h[wave][m * 64 + t * 16 + n15]        = (_Float16)softplusf(h0[t][r] + bias);
            lds_h[wave][(16 + m) * 64 + t * 16 + n15] = (_Float16)softplusf(h1[t][r] + bias);
        }
    }
    __syncthreads();

    // --- layer 2: A = h (2 M-tiles x 2 K-frags), B = w2 (64x4 padded N=16) --
    v16h b2f[2];
#pragma unroll
    for (int kf = 0; kf < 2; ++kf) {
        v16h bb = {};
        if (n15 < 4) {
            const int kb = kf * 32 + (lane < 16 ? 0 : 16);
#pragma unroll
            for (int j = 0; j < 16; ++j)
                bb[j] = (_Float16)lds_wts[576 + (kb + j) * 4 + n15];
        }
        b2f[kf] = bb;
    }

    v8f outt[2];
#pragma unroll
    for (int mi = 0; mi < 2; ++mi) {
        v8f acc = {};
#pragma unroll
        for (int kf = 0; kf < 2; ++kf) {
            v16h a = {};
            const int row = mi * 16 + n15;
            const int kb  = kf * 32 + (lane < 16 ? 0 : 16);
#pragma unroll
            for (int j = 0; j < 8; ++j) {
                a[j]     = lds_h[wave][row * 64 + kb + j];
                a[8 + j] = lds_h[wave][row * 64 + kb + 8 + j];
            }
            acc = __builtin_amdgcn_wmma_f32_16x16x32_f16(false, a, false, b2f[kf],
                                                         (short)0, acc, false, false);
        }
        outt[mi] = acc;
    }

    // --- write per-point [sigma_raw, sigmoid rgb] ---------------------------
    if (n15 < 4) {
        const float bias = lds_wts[832 + n15];
#pragma unroll
        for (int mi = 0; mi < 2; ++mi) {
#pragma unroll
            for (int r = 0; r < 8; ++r) {
                const int ml = mi * 16 + r + 8 * hi;
                const int pp = tile * 32 + ml;
                if (pp < P) {
                    float v = outt[mi][r] + bias;
                    if (n15 > 0) v = sigmoidf_(v) * 1.002f - 0.001f;
                    rgba[(size_t)pp * 4 + n15] = v;
                }
            }
        }
    }
    if (mode == 0 && valid) depths[p] = depth;
}

// ---------------------------------------------------------------------------
// Kernel 2: coarse ray-march -> weights (one thread per ray)
// ---------------------------------------------------------------------------
__global__ __launch_bounds__(256)
void march_coarse(const float* __restrict__ depths, const float* __restrict__ rgba,
                  float* __restrict__ weights, int BR, int S)
{
    const int t = blockIdx.x * blockDim.x + threadIdx.x;
    if (t >= BR) return;
    const float* z  = depths + (size_t)t * S;
    const float* rg = rgba   + (size_t)t * S * 4;
    float T = 1.f;
    for (int i = 0; i < S - 1; ++i) {
        const float dmid  = 0.5f * (rg[i * 4] + rg[(i + 1) * 4]);
        const float dens  = softplusf(dmid - 1.f);
        const float alpha = 1.f - expf(-dens * (z[i + 1] - z[i]));
        weights[(size_t)t * (S - 1) + i] = alpha * T;
        T *= (1.f - alpha + 1e-10f);
    }
}

// ---------------------------------------------------------------------------
// Kernel 3: importance sampling (inverse-CDF) -> fine depths
// ---------------------------------------------------------------------------
__global__ __launch_bounds__(256)
void importance(const float* __restrict__ depths, const float* __restrict__ weights,
                float* __restrict__ fine, int BR, int S, int NI)
{
    const int t = blockIdx.x * blockDim.x + threadIdx.x;
    if (t >= BR) return;
    float z[64], w[64], wmax[64], wp[64], zmid[64], cdf[64];
    for (int i = 0; i < S; ++i)     z[i] = depths[(size_t)t * S + i];
    for (int i = 0; i < S - 1; ++i) w[i] = weights[(size_t)t * (S - 1) + i];
    // pad +/- inf then pairwise max -> S values
    wmax[0]     = w[0];
    wmax[S - 1] = w[S - 2];
    for (int i = 1; i < S - 1; ++i) wmax[i] = fmaxf(w[i - 1], w[i]);
    for (int i = 0; i < S - 1; ++i) {
        wp[i]   = 0.5f * (wmax[i] + wmax[i + 1]) + 0.01f;
        zmid[i] = 0.5f * (z[i] + z[i + 1]);
    }
    // sample_pdf(bins=zmid[S-1], weights=wp[1..S-3] (S-3 vals), NI samples)
    const int NW = S - 3;           // 45
    float sum = 0.f;
    for (int i = 0; i < NW; ++i) { wp[1 + i] += 1e-5f; sum += wp[1 + i]; }
    cdf[0] = 0.f;
    for (int i = 0; i < NW; ++i) cdf[i + 1] = cdf[i] + wp[1 + i] / sum;
    const int NC = NW + 1;          // 46
    for (int k = 0; k < NI; ++k) {
        const float u = (float)k / (float)(NI - 1);
        int idx = 0;
        while (idx < NC && cdf[idx] <= u) ++idx;   // searchsorted right
        int below = idx - 1; if (below < 0) below = 0; if (below > NC - 1) below = NC - 1;
        int above = idx;                            if (above > NC - 1) above = NC - 1;
        const float c0 = cdf[below], c1 = cdf[above];
        const float b0 = zmid[below], b1v = zmid[above];
        const float den = (c1 - c0 < 1e-5f) ? 1.f : (c1 - c0);
        const float tt  = (u - c0) / den;
        fine[(size_t)t * NI + k] = b0 + tt * (b1v - b0);
    }
}

// ---------------------------------------------------------------------------
// Kernel 5: merge sorted coarse+fine, final ray-march -> rgb, depth, weights
// ---------------------------------------------------------------------------
__global__ __launch_bounds__(256)
void final_march(const float* __restrict__ zc, const float* __restrict__ rc,
                 const float* __restrict__ zf, const float* __restrict__ rf,
                 float* __restrict__ out, int BR, int S, int NI)
{
    const int t = blockIdx.x * blockDim.x + threadIdx.x;
    if (t >= BR) return;
    const float* zca = zc + (size_t)t * S;
    const float* zfa = zf + (size_t)t * NI;
    const float* rca = rc + (size_t)t * S * 4;
    const float* rfa = rf + (size_t)t * NI * 4;
    const int    M   = S + NI;
    float* outw = out + (size_t)BR * 4 + (size_t)t * (M - 1);

    int i = 0, j = 0;
    float T = 1.f, accR = 0.f, accG = 0.f, accB = 0.f, accD = 0.f;
    float pz = 0.f, ps = 0.f, pr = 0.f, pg = 0.f, pb = 0.f;
    for (int k = 0; k < M; ++k) {
        const bool takeC = (j >= NI) || (i < S && zca[i] <= zfa[j]);  // stable: coarse first
        float zz; const float* r4;
        if (takeC) { zz = zca[i]; r4 = rca + (size_t)i * 4; ++i; }
        else       { zz = zfa[j]; r4 = rfa + (size_t)j * 4; ++j; }
        const float sg = r4[0], rr = r4[1], gg = r4[2], bb = r4[3];
        if (k > 0) {
            const float dens  = softplusf(0.5f * (ps + sg) - 1.f);
            const float alpha = 1.f - expf(-dens * (zz - pz));
            const float wgt   = alpha * T;
            T *= (1.f - alpha + 1e-10f);
            accR += wgt * 0.5f * (pr + rr);
            accG += wgt * 0.5f * (pg + gg);
            accB += wgt * 0.5f * (pb + bb);
            accD += wgt * 0.5f * (pz + zz);
            outw[k - 1] = wgt;
        }
        pz = zz; ps = sg; pr = rr; pg = gg; pb = bb;
    }
    out[(size_t)t * 3 + 0] = accR * 2.f - 1.f;
    out[(size_t)t * 3 + 1] = accG * 2.f - 1.f;
    out[(size_t)t * 3 + 2] = accB * 2.f - 1.f;
    out[(size_t)BR * 3 + t] = accD;
}

// ---------------------------------------------------------------------------
extern "C" void kernel_launch(void* const* d_in, const int* in_sizes, int n_in,
                              void* d_out, int out_size, void* d_ws, size_t ws_size,
                              hipStream_t stream)
{
    (void)n_in; (void)out_size; (void)ws_size;
    const float* planes = (const float*)d_in[0];
    const float* w1     = (const float*)d_in[1];
    const float* b1     = (const float*)d_in[2];
    const float* w2     = (const float*)d_in[3];
    const float* b2     = (const float*)d_in[4];
    const float* ro     = (const float*)d_in[5];
    const float* rd     = (const float*)d_in[6];
    const float* noise  = (const float*)d_in[7];

    const int BR = in_sizes[5] / 3;                       // B*R
    const int S  = in_sizes[7] / BR;                      // depth_resolution (48)
    const int B  = in_sizes[0] / (3 * 32 * 256 * 256);
    const int R  = BR / B;
    const int NI = 48;                                    // depth_resolution_importance
    const int Pc = BR * S, Pf = BR * NI;

    float* ws     = (float*)d_ws;
    float* dep_c  = ws;                                   // Pc
    float* rgba_c = dep_c  + (size_t)Pc;                  // Pc*4
    float* wts    = rgba_c + (size_t)Pc * 4;              // BR*(S-1)
    float* dep_f  = wts    + (size_t)BR * (S - 1);        // Pf
    float* rgba_f = dep_f  + (size_t)Pf;                  // Pf*4

    shade_points<<<(Pc + 255) / 256, 256, 0, stream>>>(planes, w1, b1, w2, b2, ro, rd,
                                                       noise, dep_c, rgba_c, 0, Pc, R, S);
    march_coarse<<<(BR + 255) / 256, 256, 0, stream>>>(dep_c, rgba_c, wts, BR, S);
    importance<<<(BR + 255) / 256, 256, 0, stream>>>(dep_c, wts, dep_f, BR, S, NI);
    shade_points<<<(Pf + 255) / 256, 256, 0, stream>>>(planes, w1, b1, w2, b2, ro, rd,
                                                       noise, dep_f, rgba_f, 1, Pf, R, NI);
    final_march<<<(BR + 255) / 256, 256, 0, stream>>>(dep_c, rgba_c, dep_f, rgba_f,
                                                      (float*)d_out, BR, S, NI);
}
